// HeteroGNNEdgeHead_24945170055784
// MI455X (gfx1250) — compile-verified, matrix operand order checked by reading, not run on hardware
//
#include <hip/hip_runtime.h>
#include <hip/hip_bf16.h>

typedef __attribute__((ext_vector_type(16))) _Float16 v16h;
typedef __attribute__((ext_vector_type(8)))  _Float16 v8h;
typedef __attribute__((ext_vector_type(4)))  _Float16 v4h;
typedef __attribute__((ext_vector_type(8)))  float    v8f;

#define DIM_IN 128
#define D3 384               // 3*DIM_IN
#define NT 24                // 384/16 n-tiles
#define KT 12                // 384/32 k-steps
#define TILE_ELEMS 512       // 32x16 f16 per B tile
#define EDGES_PER_WAVE 64

// ---------------------------------------------------------------------------
// Pack W1 (f32 [384][384] row-major, element W1[k][j]) into f16 B-matrix
// fragment layout: tile (ntile,ktile) holds 32(K) x 16(N); per-lane 16 f16,
// lane = n + 16*((kk>>3)&1), elem = (kk&7) + 8*(kk>=16)   (ISA 7.12.2)
// ---------------------------------------------------------------------------
__global__ void pack_w1_kernel(const float* __restrict__ W1,
                               _Float16* __restrict__ W1h) {
  int tid = blockIdx.x * 256 + threadIdx.x;
  if (tid >= D3 * D3) return;
  int k = tid / D3;          // row of W1 (K dim)
  int j = tid % D3;          // col of W1 (N dim)
  int nt = j >> 4, n = j & 15;
  int kt = k >> 5, kk = k & 31;
  int lane = n + (((kk >> 3) & 1) << 4);
  int elem = (kk & 7) + ((kk >= 16) ? 8 : 0);
  size_t dst = ((size_t)(nt * KT + kt)) * TILE_ELEMS + (size_t)lane * 16 + elem;
  W1h[dst] = (_Float16)W1[tid];
}

// ---------------------------------------------------------------------------
// Main edge-MLP kernel: one wave per block, 64 edges per wave.
// ---------------------------------------------------------------------------
__global__ __launch_bounds__(32) void edge_mlp_kernel(
    const float* __restrict__ x_src, const float* __restrict__ x_dst,
    const float* __restrict__ edge_attr,
    const int* __restrict__ src_idx, const int* __restrict__ dst_idx,
    const _Float16* __restrict__ W1h,
    const float* __restrict__ b1, const float* __restrict__ W2,
    const float* __restrict__ b2, float* __restrict__ out, int E) {

  __shared__ __align__(16) _Float16 h_lds[EDGES_PER_WAVE][D3];  // 48 KB
  __shared__ __align__(16) float red[32][8];                    // 1 KB

  const int lane = threadIdx.x;
  const int tileBase = blockIdx.x * EDGES_PER_WAVE;

  // --- preload edge indices for this wave's 64 edges (2 per lane) ---
  int eA = tileBase + lane;
  int eB = eA + 32;
  int eAc = min(eA, E - 1);
  int eBc = min(eB, E - 1);
  int sA = src_idx[eAc], dA = dst_idx[eAc];
  int sB = src_idx[eBc], dB = dst_idx[eBc];

  // --- cooperative gather: 3 x 128-float rows per edge, f32 -> f16 -> LDS ---
  for (int e = 0; e < EDGES_PER_WAVE; ++e) {
    int ec = min(tileBase + e, E - 1);
    int s, d;
    if (e < 32) { s = __shfl(sA, e, 32);      d = __shfl(dA, e, 32); }
    else        { s = __shfl(sB, e - 32, 32); d = __shfl(dB, e - 32, 32); }
    float4 vs = ((const float4*)(x_src    + (size_t)s  * DIM_IN))[lane];
    float4 vd = ((const float4*)(x_dst    + (size_t)d  * DIM_IN))[lane];
    float4 ve = ((const float4*)(edge_attr + (size_t)ec * DIM_IN))[lane];
    v4h hs = { (_Float16)vs.x, (_Float16)vs.y, (_Float16)vs.z, (_Float16)vs.w };
    v4h hd = { (_Float16)vd.x, (_Float16)vd.y, (_Float16)vd.z, (_Float16)vd.w };
    v4h he = { (_Float16)ve.x, (_Float16)ve.y, (_Float16)ve.z, (_Float16)ve.w };
    *(v4h*)&h_lds[e][            lane * 4] = hs;
    *(v4h*)&h_lds[e][DIM_IN   +  lane * 4] = hd;
    *(v4h*)&h_lds[e][2*DIM_IN +  lane * 4] = he;
  }
  __syncthreads();

  const int lane_lo = lane & 15;          // A-row / B-col / D-col index
  const int khalf   = (lane >> 4) << 3;   // K sub-offset for upper half-wave

  // layer-2 partial accumulators (one chain per 16-edge sub-tile)
  v8f acc0 = {}, acc1 = {}, acc2 = {}, acc3 = {};

  for (int nt = 0; nt < NT; ++nt) {
    v8f c0 = {}, c1 = {}, c2 = {}, c3 = {};
    const _Float16* wb = W1h + (size_t)nt * KT * TILE_ELEMS;

    // Prefetch the *next* n-tile's entire 12 KB B region (32 lanes x 384 B)
    // one full iteration (48 WMMAs) ahead, at WGP scope (scope field 0 =
    // default): only scope-0 prefetch actually pulls into the WGP-local
    // cache; __builtin_prefetch maxes
    // out at SCOPE_SE which stops at GL2, where W1h already lives. The
    // addresses are always valid (inside the packed 288 KB W1h buffer), so
    // the non-speculative treatment of WGP-scope prefetch is safe.
    {
      int ntn = (nt + 1 < NT) ? (nt + 1) : 0;
      const char* p = (const char*)(W1h + (size_t)ntn * KT * TILE_ELEMS)
                      + (size_t)lane * 384;
      asm volatile("global_prefetch_b8 %0, off\n\t"
                   "global_prefetch_b8 %0, off offset:128\n\t"
                   "global_prefetch_b8 %0, off offset:256"
                   :: "v"(p));
    }

    #pragma unroll
    for (int kt = 0; kt < KT; ++kt) {
      // B fragment: 32 bytes per lane, packed by pack_w1_kernel
      v16h bm = *(const v16h*)(wb + (size_t)kt * TILE_ELEMS + (size_t)lane * 16);
      int k0 = kt * 32 + khalf;

      v8h lo0 = *(const v8h*)&h_lds[ 0 + lane_lo][k0];
      v8h hi0 = *(const v8h*)&h_lds[ 0 + lane_lo][k0 + 16];
      v8h lo1 = *(const v8h*)&h_lds[16 + lane_lo][k0];
      v8h hi1 = *(const v8h*)&h_lds[16 + lane_lo][k0 + 16];
      v8h lo2 = *(const v8h*)&h_lds[32 + lane_lo][k0];
      v8h hi2 = *(const v8h*)&h_lds[32 + lane_lo][k0 + 16];
      v8h lo3 = *(const v8h*)&h_lds[48 + lane_lo][k0];
      v8h hi3 = *(const v8h*)&h_lds[48 + lane_lo][k0 + 16];
      v16h a0 = __builtin_shufflevector(lo0, hi0, 0,1,2,3,4,5,6,7,8,9,10,11,12,13,14,15);
      v16h a1 = __builtin_shufflevector(lo1, hi1, 0,1,2,3,4,5,6,7,8,9,10,11,12,13,14,15);
      v16h a2 = __builtin_shufflevector(lo2, hi2, 0,1,2,3,4,5,6,7,8,9,10,11,12,13,14,15);
      v16h a3 = __builtin_shufflevector(lo3, hi3, 0,1,2,3,4,5,6,7,8,9,10,11,12,13,14,15);

      c0 = __builtin_amdgcn_wmma_f32_16x16x32_f16(false, a0, false, bm, (short)0, c0, false, false);
      c1 = __builtin_amdgcn_wmma_f32_16x16x32_f16(false, a1, false, bm, (short)0, c1, false, false);
      c2 = __builtin_amdgcn_wmma_f32_16x16x32_f16(false, a2, false, bm, (short)0, c2, false, false);
      c3 = __builtin_amdgcn_wmma_f32_16x16x32_f16(false, a3, false, bm, (short)0, c3, false, false);
    }

    // bias + ReLU + layer-2 (DIM_OUT = 1) folded per lane
    float b1j = b1[nt * 16 + lane_lo];
    float w2j = W2[nt * 16 + lane_lo];
    #pragma unroll
    for (int r = 0; r < 8; ++r) {
      acc0[r] += fmaxf(c0[r] + b1j, 0.0f) * w2j;
      acc1[r] += fmaxf(c1[r] + b1j, 0.0f) * w2j;
      acc2[r] += fmaxf(c2[r] + b1j, 0.0f) * w2j;
      acc3[r] += fmaxf(c3[r] + b1j, 0.0f) * w2j;
    }
  }

  // --- cross-lane reduction: sum each edge's partials over 16 lanes ---
  float b2v = b2[0];
  #pragma unroll
  for (int t = 0; t < 4; ++t) {
    __syncthreads();
    v8f a = (t == 0) ? acc0 : (t == 1) ? acc1 : (t == 2) ? acc2 : acc3;
    #pragma unroll
    for (int r = 0; r < 8; ++r) red[lane][r] = a[r];
    __syncthreads();
    if (lane < 16) {
      // rows 0..7 live in lanes 0-15 (reg r = row); rows 8..15 in lanes 16-31
      int base = (lane < 8) ? 0 : 16;
      int col  = lane & 7;
      float s = 0.0f;
      #pragma unroll
      for (int l = 0; l < 16; ++l) s += red[base + l][col];
      int eo = tileBase + t * 16 + lane;
      if (eo < E) out[eo] = s + b2v;
    }
  }
}

extern "C" void kernel_launch(void* const* d_in, const int* in_sizes, int n_in,
                              void* d_out, int out_size, void* d_ws, size_t ws_size,
                              hipStream_t stream) {
  const float* x_src     = (const float*)d_in[0];
  const float* x_dst     = (const float*)d_in[1];
  const float* edge_attr = (const float*)d_in[2];
  const int*   edge_idx  = (const int*)d_in[3];   // int32 [2][E]
  const float* W1        = (const float*)d_in[4];
  const float* b1        = (const float*)d_in[5];
  const float* W2        = (const float*)d_in[6];
  const float* b2        = (const float*)d_in[7];
  float* out = (float*)d_out;

  int E = in_sizes[3] / 2;
  _Float16* W1h = (_Float16*)d_ws;   // 384*384*2 = 294,912 bytes

  int packThreads = D3 * D3;
  pack_w1_kernel<<<(packThreads + 255) / 256, 256, 0, stream>>>(W1, W1h);

  int nBlocks = (E + EDGES_PER_WAVE - 1) / EDGES_PER_WAVE;
  edge_mlp_kernel<<<nBlocks, 32, 0, stream>>>(
      x_src, x_dst, edge_attr, edge_idx, edge_idx + E,
      W1h, b1, W2, b2, out, E);
}
